// SketchEmbed_27127013441624
// MI455X (gfx1250) — matrix operand order repaired and checked by reading, not compiled, not run on hardware
//
#include <hip/hip_runtime.h>

// ---------------------------------------------------------------------------
// GATv2 triple-encoder + MLP backbone for MI455X (gfx1250, wave32, WMMA).
//  - GEMMs: v_wmma_f32_16x16x32_bf16, B staged to LDS via async-to-LDS copies,
//    K templated (128/256) -> fully unrolled straight-line WMMA pipeline.
//  - Edge phase: CSR-by-dst built once per graph, then ONE fused per-node
//    online-softmax kernel (no atomics on features, xl[src] read once/edge).
// ---------------------------------------------------------------------------

typedef __attribute__((ext_vector_type(16))) __bf16 v16bf;
typedef __attribute__((ext_vector_type(8)))  float  v8f;
typedef __attribute__((ext_vector_type(4)))  int    v4i_;

#define NN 8192                 // nodes
#define EE 131072               // original edges
#define ET (EE + NN)            // edges + self loops
#define D  256                  // hidden/embed dim (all layers out 256)

#if defined(__AMDGCN__) && __has_builtin(__builtin_amdgcn_global_load_async_to_lds_b128) && __has_builtin(__builtin_amdgcn_s_wait_asynccnt)
#define HAVE_ASYNC_LDS 1
typedef __attribute__((address_space(1))) v4i_ gv4i;   // global int4
typedef __attribute__((address_space(3))) v4i_ lv4i;   // LDS int4
#else
#define HAVE_ASYNC_LDS 0
#endif

union FragB32 { uint4 u[2]; v16bf v; };   // 32 bytes = 16 bf16

__device__ __forceinline__ __bf16 f2bf(float f) {
  unsigned u = __float_as_uint(f);
  u = u + 0x7fffu + ((u >> 16) & 1u);            // round to nearest even
  unsigned short h = (unsigned short)(u >> 16);
  return __builtin_bit_cast(__bf16, h);
}

// --------------------------- conversion kernels ----------------------------

__global__ void cvt_kernel(const float* __restrict__ X, __bf16* __restrict__ XB, int n) {
  int i = blockIdx.x * 256 + threadIdx.x;
  if (i < n) XB[i] = f2bf(X[i]);
}

// W: [K][256] fp32 row-major  ->  WT: [256][K] bf16 (transposed)
__global__ void transpose_cvt_kernel(const float* __restrict__ W, __bf16* __restrict__ WT, int K) {
  int idx = blockIdx.x * 256 + threadIdx.x;   // grid = K blocks of 256
  int k = idx >> 8;
  int n = idx & 255;
  WT[(size_t)n * K + k] = f2bf(W[idx]);
}

// ----------------------------- CSR build (per graph) -----------------------

__global__ void deg_init_kernel(int* __restrict__ cnt) {
  cnt[blockIdx.x * 256 + threadIdx.x] = 0;      // grid = NN/256
}

__global__ void deg_count_kernel(const int* __restrict__ ei, int* __restrict__ cnt) {
  int e = blockIdx.x * 256 + threadIdx.x;
  if (e >= ET) return;
  int t = (e < EE) ? ei[EE + e] : (e - EE);
  atomicAdd(cnt + t, 1);
}

// exclusive scan of cnt[NN] -> rowptr[NN+1], cursor = rowptr copy. 1 block.
__global__ void __launch_bounds__(256) scan_kernel(const int* __restrict__ cnt,
                                                   int* __restrict__ rowptr,
                                                   int* __restrict__ cursor) {
  __shared__ int part[256];
  const int tid = threadIdx.x;
  const int base = tid * 32;
  int local[32];
  int s = 0;
#pragma unroll
  for (int j = 0; j < 32; ++j) { local[j] = s; s += cnt[base + j]; }
  part[tid] = s;
  __syncthreads();
  for (int off = 1; off < 256; off <<= 1) {     // Hillis-Steele inclusive scan
    int v = (tid >= off) ? part[tid - off] : 0;
    __syncthreads();
    part[tid] += v;
    __syncthreads();
  }
  const int pre = (tid == 0) ? 0 : part[tid - 1];
#pragma unroll
  for (int j = 0; j < 32; ++j) {
    int r = pre + local[j];
    rowptr[base + j] = r;
    cursor[base + j] = r;
  }
  if (tid == 255) rowptr[NN] = pre + s;         // == ET
}

__global__ void csr_fill_kernel(const int* __restrict__ ei, int* __restrict__ cursor,
                                int* __restrict__ csrsrc) {
  int e = blockIdx.x * 256 + threadIdx.x;
  if (e >= ET) return;
  int s, t;
  if (e < EE) { s = ei[e]; t = ei[EE + e]; } else { s = t = e - EE; }
  csrsrc[atomicAdd(cursor + t, 1)] = s;
}

// ------------------------------- WMMA GEMM ---------------------------------
// Y[NN x 256] = X[NN x K](bf16) @ W[K x 256] + bias, W given transposed (WT).
// Block (8 waves) computes 128(M) x 64(N); the B quarter [64][K] is staged in
// LDS with async-to-LDS b128 copies (row stride K+8 -> conflict-free b128
// fragment reads). Each wave: 16x64 strip = 4 accumulator tiles. K is a
// template constant so the k-loop is straight-line: all 4 B fragments are
// fetched into distinct registers before the 4 back-to-back WMMAs.
template <int K>
__global__ void __launch_bounds__(256) gemm_kernel(
    const __bf16* __restrict__ X, const __bf16* __restrict__ WT,
    const float* __restrict__ bias, float* __restrict__ Y) {
  extern __shared__ __bf16 Bs[];                // [64][K+8]
  constexpr int STRIDE = K + 8;
  const int lane  = threadIdx.x & 31;
  const int wid   = threadIdx.x >> 5;
  const int nbase = (blockIdx.x & 3) * 64;      // N quarter
  const int mblk  = (blockIdx.x >> 2) * 128;    // M block
  const int lrow  = lane & 15;
  const int lhi   = lane >> 4;

  // ---- stage B quarter into LDS ----
  constexpr int chunksPerRow = K >> 3;          // 16B chunks per row
  constexpr int total = 64 * chunksPerRow;
#pragma unroll
  for (int c0 = 0; c0 < total; c0 += 256) {
    const int c   = c0 + threadIdx.x;
    const int row = c / chunksPerRow;
    const int cc  = (c - row * chunksPerRow) * 8;
    const __bf16* src = WT + (size_t)(nbase + row) * K + cc;
    __bf16* dst = Bs + row * STRIDE + cc;
#if HAVE_ASYNC_LDS
    __bf16* snc = const_cast<__bf16*>(src);
    __builtin_amdgcn_global_load_async_to_lds_b128((gv4i*)snc, (lv4i*)dst, 0, 0);
#else
    *(uint4*)dst = *(const uint4*)src;
#endif
  }
#if HAVE_ASYNC_LDS
  __builtin_amdgcn_s_wait_asynccnt(0);
#endif
  __syncthreads();

  v8f acc[4] = {};

  // A 16x32 bf16 fragment: lane l<16 holds row l, K = {k0..k0+7, k0+16..k0+23};
  // lane l+16 holds K = {k0+8..15, k0+24..31}  -> two b128 loads per lane.
  const __bf16* arow = X + (size_t)(mblk + wid * 16 + lrow) * K + lhi * 8;

#pragma unroll
  for (int k0 = 0; k0 < K; k0 += 32) {
    FragB32 a;
    a.u[0] = *(const uint4*)(arow + k0);
    a.u[1] = *(const uint4*)(arow + k0 + 16);
    if (k0 + 32 < K)
      __builtin_prefetch(arow + k0 + 32, 0, 3);   // near-scope global_prefetch_b8
    FragB32 b[4];
#pragma unroll
    for (int t = 0; t < 4; ++t) {
      // B 32x16 fragment: lane n<16 -> K={k0..k0+7, k0+16..k0+23}; n+16 -> {+8..15, +24..31}
      const __bf16* bp = Bs + (t * 16 + lrow) * STRIDE + k0 + lhi * 8;
      b[t].u[0] = *(const uint4*)(bp);
      b[t].u[1] = *(const uint4*)(bp + 16);
    }
#pragma unroll
    for (int t = 0; t < 4; ++t)
      acc[t] = __builtin_amdgcn_wmma_f32_16x16x32_bf16(
          false, a.v, false, b[t].v, (short)0, acc[t], false, false);
  }

  // C layout: VGPR r, lanes 0-15 -> M=r, lanes 16-31 -> M=8+r
  const int m0 = mblk + wid * 16;
#pragma unroll
  for (int t = 0; t < 4; ++t) {
    const int col = nbase + t * 16 + lrow;
    const float bv = bias[col];
    float* yp = Y + (size_t)(m0 + lhi * 8) * D + col;
#pragma unroll
    for (int r = 0; r < 8; ++r)
      yp[(size_t)r * D] = acc[t][r] + bv;
  }
}

// --------------------- fused per-node GATv2 attention ----------------------
// One wave per node v: online softmax over incoming edges (CSR), each lane
// owns 8 features (f = lane + 32j). Reads xl[src] once per edge; no atomics.
// Epilogue fuses +bias, ReLU and bf16 conversion (layers 0..3), or the root
// row extraction (layer 4).
__global__ void __launch_bounds__(256) gat_node_kernel(
    const float* __restrict__ xl, const float* __restrict__ xr,
    const float* __restrict__ att, const float* __restrict__ bias,
    const int* __restrict__ rowptr, const int* __restrict__ csrsrc,
    __bf16* __restrict__ xb_out,          // layers 0..3 (else null)
    const int* __restrict__ root,         // layer 4 (else null)
    float* __restrict__ emb_out) {        // layer 4 (else null)
  const int lane = threadIdx.x & 31;
  const int v = blockIdx.x * 8 + (threadIdx.x >> 5);   // grid = NN/8 blocks

  float xrv[8], attv[8];
#pragma unroll
  for (int j = 0; j < 8; ++j) {
    const int f = lane + 32 * j;
    xrv[j]  = xr[(size_t)v * D + f];
    attv[j] = att[f];
  }

  float M = -INFINITY, S = 0.f;
  float acc[8] = {};
  const int b = rowptr[v];
  const int e = rowptr[v + 1];
  for (int i = b; i < e; ++i) {
    const int s = csrsrc[i];
    const float* ps = xl + (size_t)s * D;
    float xls[8];
    float p = 0.f;
#pragma unroll
    for (int j = 0; j < 8; ++j) {
      xls[j] = ps[lane + 32 * j];
      float m = xls[j] + xrv[j];
      m = (m > 0.f) ? m : 0.2f * m;      // leaky_relu, slope 0.2
      p = fmaf(m, attv[j], p);
    }
#pragma unroll
    for (int off = 16; off > 0; off >>= 1)
      p += __shfl_xor(p, off, 32);       // logit, broadcast to all lanes
    const float newM = fmaxf(M, p);
    const float sc = __expf(M - newM);   // first iter: exp(-inf)=0
    const float w  = __expf(p - newM);
    S = S * sc + w;
    M = newM;
#pragma unroll
    for (int j = 0; j < 8; ++j)
      acc[j] = fmaf(acc[j], sc, w * xls[j]);
  }

  const float inv = 1.f / S;             // self-loop guarantees e > b
  if (xb_out) {
#pragma unroll
    for (int j = 0; j < 8; ++j) {
      const int f = lane + 32 * j;
      float o = fmaf(acc[j], inv, bias[f]);
      o = (o > 0.f) ? o : 0.f;           // ReLU between layers
      xb_out[(size_t)v * D + f] = f2bf(o);
    }
  } else if (v == root[0]) {
#pragma unroll
    for (int j = 0; j < 8; ++j) {
      const int f = lane + 32 * j;
      emb_out[f] = fmaf(acc[j], inv, bias[f]);   // no ReLU on last layer
    }
  }
}

// -------------------------------- backbone ---------------------------------
__global__ void mlp_kernel(const float* __restrict__ in, const float* __restrict__ W,
                           const float* __restrict__ b, float* __restrict__ out,
                           int K, int relu) {
  int j = threadIdx.x;   // 256 outputs
  float s = b[j];
  for (int i = 0; i < K; ++i) s = fmaf(in[i], W[i * D + j], s);
  if (relu) s = fmaxf(s, 0.f);
  out[j] = s;
}

// ---------------------------------------------------------------------------

extern "C" void kernel_launch(void* const* d_in, const int* in_sizes, int n_in,
                              void* d_out, int out_size, void* d_ws, size_t ws_size,
                              hipStream_t stream) {
  (void)in_sizes; (void)n_in; (void)out_size; (void)ws_size;
  char* ws = (char*)d_ws;
  size_t off = 0;
  auto take = [&](size_t bytes) -> char* {
    char* p = ws + off;
    off += (bytes + 255) & ~(size_t)255;
    return p;
  };

  __bf16* xb     = (__bf16*)take((size_t)NN * D * sizeof(__bf16));  // 4 MiB
  __bf16* wtl    = (__bf16*)take((size_t)D * D * sizeof(__bf16));
  __bf16* wtr    = (__bf16*)take((size_t)D * D * sizeof(__bf16));
  float*  xl     = (float*)take((size_t)NN * D * 4);                // 8 MiB
  float*  xr     = (float*)take((size_t)NN * D * 4);                // 8 MiB
  int*    cnt    = (int*)take((size_t)NN * 4);
  int*    rowptr = (int*)take((size_t)(NN + 1) * 4);
  int*    cursor = (int*)take((size_t)NN * 4);
  int*    csrsrc = (int*)take((size_t)ET * 4);
  float*  emb    = (float*)take(3 * D * 4);      // [sketch, lhs, rhs] order
  float*  h1     = (float*)take(D * 4);
  float*  h2     = (float*)take(D * 4);

  // inputs: 0..2 x_{lhs,rhs,sketch}, 3..5 edge_index, 6..8 roots,
  // 9..38 lhs_params (5 layers x {Wl,bl,Wr,br,att,bias}), 39..68 rhs, 69..98 sketch,
  // 99..104 backbone {W1,b1,W2,b2,W3,b3}
  const int slot[3] = {1, 2, 0};   // embs = concat(sketch, lhs, rhs)

  for (int g = 0; g < 3; ++g) {
    const float* x0  = (const float*)d_in[g];
    const int*  ei   = (const int*)d_in[3 + g];
    const int*  root = (const int*)d_in[6 + g];
    const int   pb   = 9 + g * 30;

    // CSR by destination — reused by all 5 layers of this graph
    deg_init_kernel<<<NN / 256, 256, 0, stream>>>(cnt);
    deg_count_kernel<<<(ET + 255) / 256, 256, 0, stream>>>(ei, cnt);
    scan_kernel<<<1, 256, 0, stream>>>(cnt, rowptr, cursor);
    csr_fill_kernel<<<(ET + 255) / 256, 256, 0, stream>>>(ei, cursor, csrsrc);

    cvt_kernel<<<(NN * 128) / 256, 256, 0, stream>>>(x0, xb, NN * 128);

    for (int L = 0; L < 5; ++L) {
      const int K = (L == 0) ? 128 : 256;
      const float* Wl  = (const float*)d_in[pb + L * 6 + 0];
      const float* bl  = (const float*)d_in[pb + L * 6 + 1];
      const float* Wr  = (const float*)d_in[pb + L * 6 + 2];
      const float* br  = (const float*)d_in[pb + L * 6 + 3];
      const float* att = (const float*)d_in[pb + L * 6 + 4];
      const float* bia = (const float*)d_in[pb + L * 6 + 5];

      transpose_cvt_kernel<<<K, 256, 0, stream>>>(Wl, wtl, K);
      transpose_cvt_kernel<<<K, 256, 0, stream>>>(Wr, wtr, K);

      const size_t smem = (size_t)64 * (K + 8) * sizeof(__bf16);   // <= 33 KiB
      if (K == 128) {
        gemm_kernel<128><<<256, 256, smem, stream>>>(xb, wtl, bl, xl);
        gemm_kernel<128><<<256, 256, smem, stream>>>(xb, wtr, br, xr);
      } else {
        gemm_kernel<256><<<256, 256, smem, stream>>>(xb, wtl, bl, xl);
        gemm_kernel<256><<<256, 256, smem, stream>>>(xb, wtr, br, xr);
      }

      if (L < 4)
        gat_node_kernel<<<NN / 8, 256, 0, stream>>>(xl, xr, att, bia, rowptr, csrsrc,
                                                    xb, nullptr, nullptr);
      else
        gat_node_kernel<<<NN / 8, 256, 0, stream>>>(xl, xr, att, bia, rowptr, csrsrc,
                                                    nullptr, root, emb + slot[g] * D);
    }
  }

  const float* W1 = (const float*)d_in[99];
  const float* b1 = (const float*)d_in[100];
  const float* W2 = (const float*)d_in[101];
  const float* b2 = (const float*)d_in[102];
  const float* W3 = (const float*)d_in[103];
  const float* b3 = (const float*)d_in[104];

  mlp_kernel<<<1, 256, 0, stream>>>(emb, W1, b1, h1, 3 * D, 1);
  mlp_kernel<<<1, 256, 0, stream>>>(h1, W2, b2, h2, D, 1);
  mlp_kernel<<<1, 256, 0, stream>>>(h2, W3, b3, (float*)d_out, D, 0);
}
